// GINet_7834020348414
// MI455X (gfx1250) — compile-verified, matrix operand order checked by reading, not run on hardware
//
#include <hip/hip_runtime.h>
#include <hip/hip_bf16.h>
#include <stdint.h>

// ---------------------------------------------------------------------------
// Types for CDNA5 WMMA
// ---------------------------------------------------------------------------
typedef __attribute__((ext_vector_type(16))) __bf16 v16bf;
typedef __attribute__((ext_vector_type(4)))  __bf16 v4bf;
typedef __attribute__((ext_vector_type(8)))  float  v8f;

#define N_NODES 50000
#define N_EDGES 800000
#define N_GRAPHS 2000
#define EMB 300
#define EMB_PAD 320        // 300 -> multiple of 32
#define HID 600
#define HID_PAD 608        // 600 -> multiple of 32
#define FEAT 512
#define OUT 256
#define NLAYERS 5

#define TM 64
#define TN 128
#define TK 32

// ---------------------------------------------------------------------------
// h[n, 0:300] = bf16(x_emb1[x[n,0]] + x_emb2[x[n,1]]) ; pads zeroed
// ---------------------------------------------------------------------------
__global__ __launch_bounds__(256) void init_h_kernel(
    const int* __restrict__ x, const float* __restrict__ emb1,
    const float* __restrict__ emb2, __bf16* __restrict__ h, int N)
{
    int tid = blockIdx.x * blockDim.x + threadIdx.x;
    if (tid >= N * (EMB_PAD / 4)) return;
    int n = tid / (EMB_PAD / 4);
    int c = tid % (EMB_PAD / 4);
    int base = c * 4;
    v4bf o;
    if (base + 3 < EMB) {                 // chunks are wholly valid or wholly pad
        int x0 = x[2 * n];
        int x1 = x[2 * n + 1];
        const float* p1 = emb1 + (size_t)x0 * EMB + base;
        const float* p2 = emb2 + (size_t)x1 * EMB + base;
#pragma unroll
        for (int j = 0; j < 4; ++j) o[j] = (__bf16)(p1[j] + p2[j]);
    } else {
#pragma unroll
        for (int j = 0; j < 4; ++j) o[j] = (__bf16)0.0f;
    }
    *(v4bf*)(h + (size_t)n * EMB_PAD + base) = o;
}

// ---------------------------------------------------------------------------
// aggr[dst] += h[src] + e1[ea0] + e2[ea1]   (f32 atomics; h is bf16)
// Edges [0,E) real, [E,E+N) self loops with ea=(4,0). 75 chunks x 4 feats.
// ---------------------------------------------------------------------------
__global__ __launch_bounds__(256) void scatter_msgs_kernel(
    const __bf16* __restrict__ h,
    const int* __restrict__ src, const int* __restrict__ dst,
    const int* __restrict__ ea,
    const float* __restrict__ e1,        // [5][300] layer slice
    const float* __restrict__ e2,        // [3][300] layer slice
    float* __restrict__ aggr, int E, int ET)
{
    long long tid = (long long)blockIdx.x * blockDim.x + threadIdx.x;
    long long total = (long long)ET * (EMB / 4);
    if (tid >= total) return;
    int e = (int)(tid / (EMB / 4));
    int c = (int)(tid % (EMB / 4));
    int base = c * 4;

    int s, d, a0, a1;
    if (e < E) {
        s = src[e]; d = dst[e];
        a0 = ea[2 * e]; a1 = ea[2 * e + 1];
    } else {
        s = d = e - E; a0 = 4; a1 = 0;
    }

    v4bf hv = *(const v4bf*)(h + (size_t)s * EMB_PAD + base);   // 8B load
    const float* p1 = e1 + (size_t)a0 * EMB + base;
    const float* p2 = e2 + (size_t)a1 * EMB + base;

    float* o = aggr + (size_t)d * EMB_PAD + base;
    atomicAdd(o + 0, (float)hv[0] + p1[0] + p2[0]);
    atomicAdd(o + 1, (float)hv[1] + p1[1] + p2[1]);
    atomicAdd(o + 2, (float)hv[2] + p1[2] + p2[2]);
    atomicAdd(o + 3, (float)hv[3] + p1[3] + p2[3]);
}

// ---------------------------------------------------------------------------
// Weight prep: Wt[c][k] = bf16(W[k][c]), zero-padded to [NcolsPad][Kpad]
// ---------------------------------------------------------------------------
__global__ __launch_bounds__(256) void convert_w_kernel(
    const float* __restrict__ W, int K, int Nw,
    __bf16* __restrict__ Wt, int Kpad, int NcolsPad)
{
    int tid = blockIdx.x * blockDim.x + threadIdx.x;
    if (tid >= NcolsPad * Kpad) return;
    int c = tid / Kpad;
    int k = tid % Kpad;
    float v = (c < Nw && k < K) ? W[(size_t)k * Nw + c] : 0.0f;
    Wt[(size_t)c * Kpad + k] = (__bf16)v;
}

// ---------------------------------------------------------------------------
// WMMA GEMM, no LDS: fragments loaded straight from global (L2-resident).
//   C = act( A[M x lda] * Wt^T + bias )
//   A: f32 (AISBF=0, converted in regs) or bf16 (AISBF=1), K padded = lda
//   Wt: bf16 [NcolsPad x lda], N-major, zero padded -> no guards in loop
//   C: bf16 (OUTBF=1) or f32, [M x ldc]; cols in [Nw, ldc) written as zero
// Block: 256 threads (8 waves), tile 64x128; wave: 16x64 strip, 4 accumulators.
// AISBF/OUTBF are template params so the inner loop is branch-free.
// ---------------------------------------------------------------------------
template <int AISBF, int OUTBF>
__global__ __launch_bounds__(256) void gemm_wmma_kernel(
    const float* __restrict__ Af32, const __bf16* __restrict__ Abf,
    int lda,
    const __bf16* __restrict__ Wt,
    const float* __restrict__ bias, int Nw,
    float* __restrict__ Cf32, __bf16* __restrict__ Cbf, int ldc,
    int M, int relu)
{
    const int lane  = threadIdx.x & 31;
    const int wid   = threadIdx.x >> 5;
    const int wm    = (wid >> 1) * 16;     // 0,16,32,48
    const int wn    = (wid & 1) * 64;      // 0,64
    const int m0    = blockIdx.y * TM;
    const int n0    = blockIdx.x * TN;
    const int lrow  = lane & 15;
    const int khalf = (lane >> 4) * 16;    // per-lane K-half of the 32-step

    int arow = m0 + wm + lrow;
    int arc  = (arow < M) ? arow : (M - 1);   // clamp: bad rows die at store guard

    const __bf16* wbase = Wt + (size_t)(n0 + wn + lrow) * lda + khalf;
    const __bf16* abase = Abf  + (size_t)arc * lda + khalf;
    const float*  fbase = Af32 + (size_t)arc * lda + khalf;

    v8f acc[4] = {};

#pragma unroll 2
    for (int k0 = 0; k0 < lda; k0 += TK) {
        v16bf afrag;
        if (AISBF) {
            afrag = *(const v16bf*)(abase + k0);              // 32B contiguous
        } else {
            const float4* ap = (const float4*)(fbase + k0);   // 64B contiguous
            float4 q0 = ap[0], q1 = ap[1], q2 = ap[2], q3 = ap[3];
            afrag[0]=(__bf16)q0.x; afrag[1]=(__bf16)q0.y; afrag[2]=(__bf16)q0.z; afrag[3]=(__bf16)q0.w;
            afrag[4]=(__bf16)q1.x; afrag[5]=(__bf16)q1.y; afrag[6]=(__bf16)q1.z; afrag[7]=(__bf16)q1.w;
            afrag[8]=(__bf16)q2.x; afrag[9]=(__bf16)q2.y; afrag[10]=(__bf16)q2.z; afrag[11]=(__bf16)q2.w;
            afrag[12]=(__bf16)q3.x; afrag[13]=(__bf16)q3.y; afrag[14]=(__bf16)q3.z; afrag[15]=(__bf16)q3.w;
        }
#pragma unroll
        for (int j = 0; j < 4; ++j) {
            v16bf bfrag = *(const v16bf*)(wbase + (size_t)j * 16 * lda + k0);
            acc[j] = __builtin_amdgcn_wmma_f32_16x16x32_bf16(
                false, afrag, false, bfrag, (short)0, acc[j], false, false);
        }
    }

    // epilogue: bias + activation, C layout per ISA 7.12.2
    const int rbase = m0 + wm + (lane >> 4) * 8;
#pragma unroll
    for (int j = 0; j < 4; ++j) {
        int col = n0 + wn + j * 16 + lrow;
        if (col >= ldc) continue;
        float bcol = (col < Nw) ? bias[col] : 0.0f;
#pragma unroll
        for (int r = 0; r < 8; ++r) {
            int row = rbase + r;
            if (row < M) {
                float v = acc[j][r] + bcol;
                if (col >= Nw)   v = 0.0f;       // keep K-pads of next GEMM clean
                else if (relu)   v = fmaxf(v, 0.0f);
                if (OUTBF) Cbf [(size_t)row * ldc + col] = (__bf16)v;
                else       Cf32[(size_t)row * ldc + col] = v;
            }
        }
    }
}

// ---------------------------------------------------------------------------
// Global mean pool (h is bf16, pools f32)
// ---------------------------------------------------------------------------
__global__ __launch_bounds__(256) void pool_sum_kernel(
    const __bf16* __restrict__ h, const int* __restrict__ batch,
    float* __restrict__ pools, float* __restrict__ counts, int N)
{
    int tid = blockIdx.x * blockDim.x + threadIdx.x;
    if (tid >= N * (EMB_PAD / 4)) return;
    int n = tid / (EMB_PAD / 4);
    int c = tid % (EMB_PAD / 4);
    int g = batch[n];
    v4bf v = *(const v4bf*)(h + (size_t)n * EMB_PAD + c * 4);
    float* o = pools + (size_t)g * EMB_PAD + c * 4;
    atomicAdd(o + 0, (float)v[0]);
    atomicAdd(o + 1, (float)v[1]);
    atomicAdd(o + 2, (float)v[2]);
    atomicAdd(o + 3, (float)v[3]);
    if (c == 0) atomicAdd(&counts[g], 1.0f);
}

__global__ __launch_bounds__(256) void pool_div_kernel(
    float* __restrict__ pools, const float* __restrict__ counts, int G)
{
    int tid = blockIdx.x * blockDim.x + threadIdx.x;
    if (tid >= G * EMB_PAD) return;
    int g = tid / EMB_PAD;
    pools[tid] /= fmaxf(counts[g], 1.0f);
}

// ---------------------------------------------------------------------------
// Host-side orchestration
// ---------------------------------------------------------------------------
static inline int ncols_pad(int ldc) { return ((ldc + TN - 1) / TN) * TN; }

static inline void launch_convert_w(const float* W, int K, int Nw, __bf16* Wt,
                                    int Kpad, int NcolsPad, hipStream_t stream)
{
    int work = NcolsPad * Kpad;
    convert_w_kernel<<<(work + 255) / 256, 256, 0, stream>>>(W, K, Nw, Wt, Kpad, NcolsPad);
}

static inline void launch_gemm(const float* Af32, const __bf16* Abf, int lda,
                               const __bf16* Wt, const float* bias, int Nw,
                               float* Cf32, __bf16* Cbf, int ldc,
                               int M, int relu, hipStream_t stream)
{
    dim3 grid((ldc + TN - 1) / TN, (M + TM - 1) / TM);
    if (Abf) {
        if (Cbf)
            gemm_wmma_kernel<1, 1><<<grid, 256, 0, stream>>>(
                Af32, Abf, lda, Wt, bias, Nw, Cf32, Cbf, ldc, M, relu);
        else
            gemm_wmma_kernel<1, 0><<<grid, 256, 0, stream>>>(
                Af32, Abf, lda, Wt, bias, Nw, Cf32, Cbf, ldc, M, relu);
    } else {
        if (Cbf)
            gemm_wmma_kernel<0, 1><<<grid, 256, 0, stream>>>(
                Af32, Abf, lda, Wt, bias, Nw, Cf32, Cbf, ldc, M, relu);
        else
            gemm_wmma_kernel<0, 0><<<grid, 256, 0, stream>>>(
                Af32, Abf, lda, Wt, bias, Nw, Cf32, Cbf, ldc, M, relu);
    }
}

extern "C" void kernel_launch(void* const* d_in, const int* in_sizes, int n_in,
                              void* d_out, int out_size, void* d_ws, size_t ws_size,
                              hipStream_t stream)
{
    (void)in_sizes; (void)n_in; (void)out_size; (void)ws_size;

    const int*   x       = (const int*)  d_in[0];
    const int*   eidx    = (const int*)  d_in[1];   // [2, E]
    const int*   eattr   = (const int*)  d_in[2];   // [E, 2]
    const int*   batch   = (const int*)  d_in[3];
    const float* x_emb1  = (const float*)d_in[4];
    const float* x_emb2  = (const float*)d_in[5];
    const float* edge_e1 = (const float*)d_in[6];   // [L,5,300]
    const float* edge_e2 = (const float*)d_in[7];   // [L,3,300]
    const float* W1      = (const float*)d_in[8];   // [L,300,600]
    const float* b1      = (const float*)d_in[9];
    const float* W2      = (const float*)d_in[10];  // [L,600,300]
    const float* b2      = (const float*)d_in[11];
    const float* feat_W  = (const float*)d_in[12];  // [300,512]
    const float* feat_b  = (const float*)d_in[13];
    const float* p0_W    = (const float*)d_in[14];  // [512,256]
    const float* p0_b    = (const float*)d_in[15];
    const float* p1_W    = (const float*)d_in[16];  // [256,256]
    const float* p1_b    = (const float*)d_in[17];
    const float* p2_W    = (const float*)d_in[18];  // [256,256]
    const float* p2_b    = (const float*)d_in[19];

    float* out = (float*)d_out;

    // ---- workspace carve-up (256B aligned chunks) ----
    char* wsb = (char*)d_ws;
    size_t off = 0;
    auto take = [&](size_t bytes) -> void* {
        void* p = wsb + off;
        off = (off + bytes + 255) & ~((size_t)255);
        return p;
    };
    __bf16* h      = (__bf16*)take((size_t)N_NODES * EMB_PAD * 2);   // 32.0 MB
    float*  aggr   = (float*) take((size_t)N_NODES * EMB_PAD * 4);   // 64.0 MB
    __bf16* tmp    = (__bf16*)take((size_t)N_NODES * HID_PAD * 2);   // 60.8 MB
    float*  pools  = (float*) take((size_t)N_GRAPHS * EMB_PAD * 4);
    float*  cnts   = (float*) take((size_t)N_GRAPHS * 4);
    __bf16* pbuf0  = (__bf16*)take((size_t)N_GRAPHS * OUT * 2);
    __bf16* pbuf1  = (__bf16*)take((size_t)N_GRAPHS * OUT * 2);
    __bf16* W1t    = (__bf16*)take((size_t)ncols_pad(HID_PAD) * EMB_PAD * 2); // 640x320
    __bf16* W2t    = (__bf16*)take((size_t)ncols_pad(EMB_PAD) * HID_PAD * 2); // 384x608
    __bf16* featWt = (__bf16*)take((size_t)ncols_pad(FEAT) * EMB_PAD * 2);    // 512x320
    __bf16* p0t    = (__bf16*)take((size_t)ncols_pad(OUT) * FEAT * 2);        // 256x512
    __bf16* p1t    = (__bf16*)take((size_t)ncols_pad(OUT) * OUT * 2);         // 256x256
    __bf16* p2t    = (__bf16*)take((size_t)ncols_pad(OUT) * OUT * 2);         // 256x256

    const int ET = N_EDGES + N_NODES;
    const int* src = eidx;
    const int* dst = eidx + N_EDGES;

    // ---- initial embedding ----
    {
        int work = N_NODES * (EMB_PAD / 4);
        init_h_kernel<<<(work + 255) / 256, 256, 0, stream>>>(x, x_emb1, x_emb2,
                                                              h, N_NODES);
    }

    // ---- GINEConv layers ----
    for (int l = 0; l < NLAYERS; ++l) {
        // weight prep for this layer (bf16, transposed, zero padded)
        launch_convert_w(W1 + (size_t)l * EMB * HID, EMB, HID,
                         W1t, EMB_PAD, ncols_pad(HID_PAD), stream);
        launch_convert_w(W2 + (size_t)l * HID * EMB, HID, EMB,
                         W2t, HID_PAD, ncols_pad(EMB_PAD), stream);

        hipMemsetAsync(aggr, 0, (size_t)N_NODES * EMB_PAD * sizeof(float), stream);

        long long work   = (long long)ET * (EMB / 4);
        long long blocks = (work + 255) / 256;
        scatter_msgs_kernel<<<(unsigned)blocks, 256, 0, stream>>>(
            h, src, dst, eattr,
            edge_e1 + (size_t)l * 5 * EMB,
            edge_e2 + (size_t)l * 3 * EMB,
            aggr, N_EDGES, ET);

        // tmp = relu(aggr @ W1 + b1)      f32 A path, bf16 out [N,608]
        launch_gemm(aggr, nullptr, EMB_PAD, W1t, b1 + (size_t)l * HID, HID,
                    nullptr, tmp, HID_PAD, N_NODES, 1, stream);
        // h = act(tmp @ W2 + b2)          bf16 A path, bf16 out [N,320]
        launch_gemm(nullptr, tmp, HID_PAD, W2t, b2 + (size_t)l * EMB, EMB,
                    nullptr, h, EMB_PAD, N_NODES, (l < NLAYERS - 1) ? 1 : 0,
                    stream);
    }

    // ---- global mean pool ----
    hipMemsetAsync(pools, 0, (size_t)N_GRAPHS * EMB_PAD * sizeof(float), stream);
    hipMemsetAsync(cnts, 0, (size_t)N_GRAPHS * sizeof(float), stream);
    {
        int work = N_NODES * (EMB_PAD / 4);
        pool_sum_kernel<<<(work + 255) / 256, 256, 0, stream>>>(h, batch, pools,
                                                                cnts, N_NODES);
        int work2 = N_GRAPHS * EMB_PAD;
        pool_div_kernel<<<(work2 + 255) / 256, 256, 0, stream>>>(pools, cnts,
                                                                 N_GRAPHS);
    }

    // ---- head ----
    launch_convert_w(feat_W, EMB, FEAT, featWt, EMB_PAD, ncols_pad(FEAT), stream);
    launch_convert_w(p0_W, FEAT, OUT, p0t, FEAT, ncols_pad(OUT), stream);
    launch_convert_w(p1_W, OUT, OUT, p1t, OUT, ncols_pad(OUT), stream);
    launch_convert_w(p2_W, OUT, OUT, p2t, OUT, ncols_pad(OUT), stream);

    // hg = pools @ feat_W + feat_b  -> f32 d_out[0 : G*512]
    launch_gemm(pools, nullptr, EMB_PAD, featWt, feat_b, FEAT,
                out, nullptr, FEAT, N_GRAPHS, 0, stream);
    // p = relu(hg @ p0_W + p0_b)    f32 A (d_out), bf16 out
    launch_gemm(out, nullptr, FEAT, p0t, p0_b, OUT,
                nullptr, pbuf0, OUT, N_GRAPHS, 1, stream);
    // p = relu(p @ p1_W + p1_b)
    launch_gemm(nullptr, pbuf0, OUT, p1t, p1_b, OUT,
                nullptr, pbuf1, OUT, N_GRAPHS, 1, stream);
    // proj = p @ p2_W + p2_b -> f32 d_out[G*512 : ]
    launch_gemm(nullptr, pbuf1, OUT, p2t, p2_b, OUT,
                out + (size_t)N_GRAPHS * FEAT, nullptr, OUT, N_GRAPHS, 0, stream);
}
